// DeepSetTransformer_13262859010115
// MI455X (gfx1250) — matrix operand order repaired and checked by reading, not compiled, not run on hardware
//
#include <hip/hip_runtime.h>
#include <cstdint>
#include <cstddef>

// ---------------------------------------------------------------------------
// DeepSet-Transformer forward for MI455X (gfx1250, wave32, WMMA).
// All large GEMMs run through v_wmma_f32_16x16x32_bf16 (f32 accumulate).
// Each wave computes a 64x16 output strip (4 accumulators); per k-step all
// fragments (1 B + 4 A) are loaded into distinct registers first, then the
// 4 independent WMMAs issue back-to-back so loads overlap the matrix pipe.
// ---------------------------------------------------------------------------

#define BB    128
#define NN    512
#define DIND  512
#define HIDD  256
#define NHEAD 8
#define DH    32
#define INDSN 64

typedef __attribute__((ext_vector_type(16))) __bf16 v16bf;
typedef __attribute__((ext_vector_type(8)))  float  v8f;
using u16 = unsigned short;

#define DEV __device__ __forceinline__

DEV u16 f2bf(float f) {
  union { float f; uint32_t u; } c; c.f = f;
  uint32_t x = c.u;
  return (u16)((x + 0x7FFFu + ((x >> 16) & 1u)) >> 16);  // RNE
}
DEV float bf2f(u16 h) {
  union { uint32_t u; float f; } c; c.u = (uint32_t)h << 16;
  return c.f;
}

// ----------------------------- conversion kernels --------------------------

__global__ __launch_bounds__(256)
void k_f32_to_bf16(const float* __restrict__ in, u16* __restrict__ out, size_t n) {
  size_t i = (size_t)blockIdx.x * blockDim.x + threadIdx.x;
  if (i < n) out[i] = f2bf(in[i]);
}

// in: [R,C] f32 row-major  ->  out: [C,R] bf16 row-major (i.e. W^T, ld = R)
__global__ __launch_bounds__(256)
void k_f32_to_bf16_T(const float* __restrict__ in, u16* __restrict__ out, int R, int C) {
  int i = blockIdx.x * blockDim.x + threadIdx.x;
  if (i >= R * C) return;
  int r = i / C, c = i % C;
  out[(size_t)c * R + r] = f2bf(in[i]);
}

// ----------------------------- WMMA GEMM -----------------------------------
// D[z][m][n] = epi( sum_k A[z][m][k] * W[z][n][k] + bias[n] )
// A bf16 row-major (lda), W bf16 "B^T" row-major (ldw). M % 64 == 0, K % 32 == 0.
// flags: 1 = ReLU, 2 = LeakyReLU(0.3). Epilogue order: bias -> act -> +resid.
// Optional outputs: f32 row-major, bf16 row-major, bf16 TRANSPOSED ([n][m]).

union FragU { uint4 u[2]; v16bf v; };

__global__ __launch_bounds__(128)
void k_wmma_gemm(const u16* __restrict__ A, long long sA, int lda,
                 const u16* __restrict__ W, long long sW, int ldw,
                 const float* __restrict__ bias,
                 const float* __restrict__ resid, long long sR, int ldr,
                 float* __restrict__ outF, long long sF, int ldf,
                 u16* __restrict__ outB, long long sB, int ldb,
                 u16* __restrict__ outBT, long long sBT, int ldbt,
                 int M, int N, int K, int flags)
{
  int nt = blockIdx.x * 4 + threadIdx.y;      // wave-uniform n-tile
  if (nt * 16 >= N) return;
  int m0 = blockIdx.y * 64;                   // 64-row strip per wave
  int z  = blockIdx.z;
  int lane = threadIdx.x;
  int hl = lane >> 4, r = lane & 15;
  int n0 = nt * 16;

  const u16* Arow0 = A + (size_t)z * sA + (size_t)(m0 + r) * lda;
  const u16* Arow1 = Arow0 + (size_t)16 * lda;
  const u16* Arow2 = Arow0 + (size_t)32 * lda;
  const u16* Arow3 = Arow0 + (size_t)48 * lda;
  const u16* Wrow  = W + (size_t)z * sW + (size_t)(n0 + r) * ldw;

  v8f acc0 = {}, acc1 = {}, acc2 = {}, acc3 = {};
  for (int k0 = 0; k0 < K; k0 += 32) {
    // issue ALL fragment loads first (distinct registers), then the 4 WMMAs
    FragU fb, fa0, fa1, fa2, fa3;
    fb.u[0]  = *(const uint4*)(Wrow  + k0 + hl * 8);
    fb.u[1]  = *(const uint4*)(Wrow  + k0 + 16 + hl * 8);
    fa0.u[0] = *(const uint4*)(Arow0 + k0 + hl * 8);
    fa0.u[1] = *(const uint4*)(Arow0 + k0 + 16 + hl * 8);
    fa1.u[0] = *(const uint4*)(Arow1 + k0 + hl * 8);
    fa1.u[1] = *(const uint4*)(Arow1 + k0 + 16 + hl * 8);
    fa2.u[0] = *(const uint4*)(Arow2 + k0 + hl * 8);
    fa2.u[1] = *(const uint4*)(Arow2 + k0 + 16 + hl * 8);
    fa3.u[0] = *(const uint4*)(Arow3 + k0 + hl * 8);
    fa3.u[1] = *(const uint4*)(Arow3 + k0 + 16 + hl * 8);
    if (k0 + 32 < K) __builtin_prefetch(Arow0 + k0 + 32, 0, 1);  // global_prefetch
    acc0 = __builtin_amdgcn_wmma_f32_16x16x32_bf16(false, fa0.v, false, fb.v,
                                                   (short)0, acc0, false, false);
    acc1 = __builtin_amdgcn_wmma_f32_16x16x32_bf16(false, fa1.v, false, fb.v,
                                                   (short)0, acc1, false, false);
    acc2 = __builtin_amdgcn_wmma_f32_16x16x32_bf16(false, fa2.v, false, fb.v,
                                                   (short)0, acc2, false, false);
    acc3 = __builtin_amdgcn_wmma_f32_16x16x32_bf16(false, fa3.v, false, fb.v,
                                                   (short)0, acc3, false, false);
  }

  int n = n0 + r;
  float bv = bias ? bias[n] : 0.0f;
  v8f accs[4] = {acc0, acc1, acc2, acc3};
#pragma unroll
  for (int t = 0; t < 4; ++t) {
#pragma unroll
    for (int i = 0; i < 8; ++i) {
      int m = m0 + t * 16 + i + hl * 8;
      float v = accs[t][i] + bv;
      if (flags & 1) v = v > 0.0f ? v : 0.0f;
      if (flags & 2) v = v > 0.0f ? v : 0.3f * v;
      if (resid) v += resid[(size_t)z * sR + (size_t)m * ldr + n];
      if (outF)  outF [(size_t)z * sF  + (size_t)m * ldf  + n] = v;
      if (outB)  outB [(size_t)z * sB  + (size_t)m * ldb  + n] = f2bf(v);
      if (outBT) outBT[(size_t)z * sBT + (size_t)n * ldbt + m] = f2bf(v);
    }
  }
}

// ----------------------------- softmax (wave/row) --------------------------

template <int PER>   // row length = PER*32
__global__ __launch_bounds__(32)
void k_softmax(const float* __restrict__ S, u16* __restrict__ P, float scale) {
  size_t row = blockIdx.x;
  const int nk = PER * 32;
  const float* s = S + row * (size_t)nk;
  u16* p = P + row * (size_t)nk;
  int lane = threadIdx.x;
  float vals[PER];
  float mx = -3.4e38f;
#pragma unroll
  for (int i = 0; i < PER; ++i) { vals[i] = s[lane + i * 32] * scale; mx = fmaxf(mx, vals[i]); }
  for (int o = 16; o > 0; o >>= 1) mx = fmaxf(mx, __shfl_xor(mx, o, 32));
  float sum = 0.0f;
#pragma unroll
  for (int i = 0; i < PER; ++i) { vals[i] = __expf(vals[i] - mx); sum += vals[i]; }
  for (int o = 16; o > 0; o >>= 1) sum += __shfl_xor(sum, o, 32);
  float inv = 1.0f / sum;
#pragma unroll
  for (int i = 0; i < PER; ++i) p[lane + i * 32] = f2bf(vals[i] * inv);
}

// ----------------------------- layernorm (wave/row, dim 256) ----------------

__global__ __launch_bounds__(32)
void k_layernorm(const float* __restrict__ in, const float* __restrict__ g,
                 const float* __restrict__ bta,
                 float* __restrict__ outF, u16* __restrict__ outB, int ldOutB) {
  size_t row = blockIdx.x;
  const float* x = in + row * (size_t)HIDD;
  int lane = threadIdx.x;
  float v[8];
  float s = 0.0f;
#pragma unroll
  for (int i = 0; i < 8; ++i) { v[i] = x[lane + i * 32]; s += v[i]; }
  for (int o = 16; o > 0; o >>= 1) s += __shfl_xor(s, o, 32);
  float mu = s * (1.0f / HIDD);
  float q = 0.0f;
#pragma unroll
  for (int i = 0; i < 8; ++i) { float d = v[i] - mu; q += d * d; }
  for (int o = 16; o > 0; o >>= 1) q += __shfl_xor(q, o, 32);
  float rstd = rsqrtf(q * (1.0f / HIDD) + 1e-5f);
#pragma unroll
  for (int i = 0; i < 8; ++i) {
    int c = lane + i * 32;
    float y = (v[i] - mu) * rstd * g[c] + bta[c];
    if (outF) outF[row * (size_t)HIDD + c] = y;
    if (outB) outB[row * (size_t)ldOutB + c] = f2bf(y);
  }
}

// -------------- PMA attention (nq == 1, shared q, nk = 512) -----------------
// O[b][h*32+d] = q[h*32+d] + sum_j softmax_j(q.k/16) * v[b][j][h*32+d]

__global__ __launch_bounds__(256)
void k_attn_q1(const float* __restrict__ q,       // [256], batch-shared
               const u16* __restrict__ kbf,       // [B, nk, 256] bf16
               const u16* __restrict__ vT,        // [B, 256, nk] bf16
               float* __restrict__ O,             // [B, 256] f32
               int nk, float scale) {
  int b = blockIdx.x, h = blockIdx.y;
  __shared__ float sp[512];
  __shared__ float red[256];
  int tid = threadIdx.x;
  const u16* kb = kbf + (size_t)b * nk * HIDD + h * DH;
  for (int j = tid; j < nk; j += 256) {
    const u16* kr = kb + (size_t)j * HIDD;
    float acc = 0.0f;
#pragma unroll
    for (int d = 0; d < DH; ++d) acc += q[h * DH + d] * bf2f(kr[d]);
    sp[j] = acc * scale;
  }
  __syncthreads();
  float mx = -3.4e38f;
  for (int j = tid; j < nk; j += 256) mx = fmaxf(mx, sp[j]);
  red[tid] = mx; __syncthreads();
  for (int s = 128; s > 0; s >>= 1) {
    if (tid < s) red[tid] = fmaxf(red[tid], red[tid + s]);
    __syncthreads();
  }
  mx = red[0]; __syncthreads();
  float sum = 0.0f;
  for (int j = tid; j < nk; j += 256) { float e = __expf(sp[j] - mx); sp[j] = e; sum += e; }
  red[tid] = sum; __syncthreads();
  for (int s = 128; s > 0; s >>= 1) {
    if (tid < s) red[tid] += red[tid + s];
    __syncthreads();
  }
  float inv = 1.0f / red[0];
  if (tid < DH) {
    int d = tid;
    const u16* vr = vT + (size_t)b * HIDD * nk + (size_t)(h * DH + d) * nk;
    float acc = 0.0f;
    for (int j = 0; j < nk; ++j) acc += sp[j] * bf2f(vr[j]);
    O[(size_t)b * HIDD + h * DH + d] = q[h * DH + d] + acc * inv;
  }
}

// -------------- scalar row-GEMM for tiny shapes (nq==1 paths, head) ---------
// out[row][n] = epi( sum_k A[row][k]*W[k][n] + bias[n] )   (W is original f32)

__global__ __launch_bounds__(64)
void k_rowgemm(const float* __restrict__ A, long long strideA,
               const float* __restrict__ W, const float* __restrict__ bias,
               const float* __restrict__ resid, long long strideR,
               float* __restrict__ outF, long long strideO,
               int K, int N, int flags) {
  int row = blockIdx.x;
  int n = blockIdx.y * 64 + threadIdx.x;
  if (n >= N) return;
  const float* a = A + (size_t)row * strideA;
  float acc = bias ? bias[n] : 0.0f;
  for (int k = 0; k < K; ++k) acc += a[k] * W[(size_t)k * N + n];
  if (flags & 1) acc = fmaxf(acc, 0.0f);
  if (flags & 2) acc = acc > 0.0f ? acc : 0.3f * acc;
  if (resid) acc += resid[(size_t)row * strideR + n];
  outF[(size_t)row * strideO + n] = acc;
}

__global__ __launch_bounds__(256)
void k_add(const float* __restrict__ a, const float* __restrict__ b,
           float* __restrict__ out, size_t n) {
  size_t i = (size_t)blockIdx.x * blockDim.x + threadIdx.x;
  if (i < n) out[i] = a[i] + b[i];
}

// mean over N of in[B,N,C], write bf16 into out[b*ldOut + c]
__global__ __launch_bounds__(256)
void k_mean(const float* __restrict__ in, int Nn, int C,
            u16* __restrict__ outB, int ldOut) {
  int b = blockIdx.x, c = threadIdx.x;
  const float* p = in + (size_t)b * Nn * C + c;
  float s = 0.0f;
  for (int j = 0; j < Nn; ++j) s += p[(size_t)j * C];
  outB[(size_t)b * ldOut + c] = f2bf(s / (float)Nn);
}

// ---------------------------------------------------------------------------

extern "C" void kernel_launch(void* const* d_in, const int* in_sizes, int n_in,
                              void* d_out, int out_size, void* d_ws, size_t ws_size,
                              hipStream_t stream)
{
  (void)in_sizes; (void)n_in; (void)out_size; (void)ws_size;

  // ---- parse params (setup_inputs() insertion order, recursively) ----
  int idx = 0;
  auto nextp = [&]() { return (const float*)d_in[idx++]; };
  struct LinP { const float *w, *b; };
  struct LnP  { const float *g, *b; };
  struct MabP { LinP q, k, v, o; LnP ln0, ln1; };
  auto lin  = [&]() { LinP l; l.w = nextp(); l.b = nextp(); return l; };
  auto lnp  = [&]() { LnP l; l.g = nextp(); l.b = nextp(); return l; };
  auto mabp = [&]() { MabP m; m.q = lin(); m.k = lin(); m.v = lin(); m.o = lin();
                      m.ln0 = lnp(); m.ln1 = lnp(); return m; };

  const float* x   = nextp();
  const float* I0  = nextp(); MabP m00 = mabp(); MabP m01 = mabp();
  const float* I1  = nextp(); MabP m10 = mabp(); MabP m11 = mabp();
  const float* Ssd = nextp(); MabP mpm = mabp();
  MabP ms0 = mabp(); MabP ms1 = mabp();
  LinP ds0 = lin(), ds1 = lin(), h0 = lin(), h1 = lin();

  // ---- workspace bump allocator ----
  char* wsb = (char*)d_ws; size_t cur = 0;
  auto alloc = [&](size_t bytes) -> void* {
    void* p = wsb + cur;
    cur += (bytes + 255) & ~(size_t)255;
    return p;
  };

  const size_t E1 = (size_t)BB * NN * HIDD;           // 16.78M
  const size_t E2 = (size_t)BB * INDSN * HIDD;        // 2.10M
  const size_t ES = (size_t)BB * NHEAD * INDSN * NN;  // 33.55M (score elems)

  auto bfcopy = [&](const float* src, size_t n) -> u16* {
    u16* dst = (u16*)alloc(n * 2);
    k_f32_to_bf16<<<dim3((unsigned)((n + 255) / 256)), 256, 0, stream>>>(src, dst, n);
    return dst;
  };
  auto bfT = [&](const float* src, int K, int N) -> u16* {
    u16* dst = (u16*)alloc((size_t)K * N * 2);
    int tot = K * N;
    k_f32_to_bf16_T<<<dim3((tot + 255) / 256), 256, 0, stream>>>(src, dst, K, N);
    return dst;
  };

  // transposed bf16 weights for all WMMA-consumed matrices
  struct TW { const u16 *q, *k, *v, *o; };
  auto twFor = [&](const MabP& m, int dq, int dk) {
    TW t; t.q = bfT(m.q.w, dq, HIDD); t.k = bfT(m.k.w, dk, HIDD);
    t.v = bfT(m.v.w, dk, HIDD); t.o = bfT(m.o.w, HIDD, HIDD); return t;
  };
  TW t00 = twFor(m00, HIDD, DIND);
  TW t01 = twFor(m01, DIND, HIDD);
  TW t10 = twFor(m10, HIDD, HIDD);
  TW t11 = twFor(m11, HIDD, HIDD);
  const u16* pkT = bfT(mpm.k.w, HIDD, HIDD);
  const u16* pvT = bfT(mpm.v.w, HIDD, HIDD);
  const u16* d0T = bfT(ds0.w, DIND, HIDD);
  const u16* d1T = bfT(ds1.w, HIDD, HIDD);
  const u16* h0T = bfT(h0.w, 2 * HIDD, HIDD / 2);

  // bf16 activations
  u16* x_bf  = bfcopy(x,  (size_t)BB * NN * DIND);
  u16* I0_bf = bfcopy(I0, (size_t)INDSN * HIDD);
  u16* I1_bf = bfcopy(I1, (size_t)INDSN * HIDD);

  // big reusable scratch
  float* q_f   = (float*)alloc(E1 * 4);
  u16*   q_bf  = (u16*)  alloc(E1 * 2);
  u16*   k_bf  = (u16*)  alloc(E1 * 2);
  u16*   vT_bf = (u16*)  alloc(E1 * 2);
  float* Sc    = (float*)alloc(ES * 4);
  u16*   Pr    = (u16*)  alloc(ES * 2);
  float* Ob    = (float*)alloc(E1 * 4);
  float* OlnF  = (float*)alloc(E1 * 4);
  u16*   OlnB  = (u16*)  alloc(E1 * 2);
  float* O2    = (float*)alloc(E1 * 4);
  u16*   Hbf   = (u16*)  alloc(E2 * 2);
  u16*   tA    = (u16*)  alloc(E1 * 2);
  u16*   tB    = (u16*)  alloc(E1 * 2);
  // small buffers
  float* qS  = (float*)alloc((size_t)HIDD * 4);
  float* Op  = (float*)alloc((size_t)BB * HIDD * 4);
  float* OpL = (float*)alloc((size_t)BB * HIDD * 4);
  float* Op2 = (float*)alloc((size_t)BB * HIDD * 4);
  float* tp  = (float*)alloc((size_t)BB * HIDD * 4);
  float* qv  = (float*)alloc((size_t)BB * HIDD * 4);
  float* vv  = (float*)alloc((size_t)BB * HIDD * 4);
  float* Os  = (float*)alloc((size_t)BB * HIDD * 4);
  float* OsL = (float*)alloc((size_t)BB * HIDD * 4);
  float* Os2 = (float*)alloc((size_t)BB * HIDD * 4);
  float* ts0 = (float*)alloc((size_t)BB * HIDD * 4);
  u16*   hcat= (u16*)  alloc((size_t)BB * 2 * HIDD * 2);
  float* h0f = (float*)alloc((size_t)BB * (HIDD / 2) * 4);

  auto gemm = [&](const u16* A, long long sA, int lda,
                  const u16* W, long long sW, int ldw,
                  const float* bias,
                  const float* resid, long long sR, int ldr,
                  float* oF, long long sF, int ldf,
                  u16* oB, long long sB2, int ldb,
                  u16* oBT, long long sBT, int ldbt,
                  int M, int N, int K, int Z, int flags) {
    dim3 g((N + 63) / 64, M / 64, Z), b(32, 4);   // 64-row strip per wave
    k_wmma_gemm<<<g, b, 0, stream>>>(A, sA, lda, W, sW, ldw, bias, resid, sR, ldr,
                                     oF, sF, ldf, oB, sB2, ldb, oBT, sBT, ldbt,
                                     M, N, K, flags);
  };

  // ---- full MAB with WMMA attention (nq in {64,512}) ----
  auto mab_big = [&](const MabP& m, const TW& tw,
                     const u16* Qsrc, int nq, int dq, bool qShared,
                     const u16* Ksrc, int nk, int dk,
                     u16* out_bf, float* out_f) {
    const int V = HIDD, H = NHEAD;
    int qRows = qShared ? nq : BB * nq;
    // projections (flattened over batch where possible)
    gemm(Qsrc, 0, dq, tw.q, 0, dq, m.q.b, nullptr, 0, 0,
         q_f, 0, V, q_bf, 0, V, nullptr, 0, 0, qRows, V, dq, 1, 0);
    gemm(Ksrc, 0, dk, tw.k, 0, dk, m.k.b, nullptr, 0, 0,
         nullptr, 0, 0, k_bf, 0, V, nullptr, 0, 0, BB * nk, V, dk, 1, 0);
    // v projection written TRANSPOSED per batch: vT[b][c][j]
    gemm(Ksrc, (long long)nk * dk, dk, tw.v, 0, dk, m.v.b, nullptr, 0, 0,
         nullptr, 0, 0, nullptr, 0, 0, vT_bf, (long long)V * nk, nk,
         nk, V, dk, BB, 0);
    // scores: per head, K = 32, A = q slice, B^T = k slice
    for (int h = 0; h < H; ++h)
      gemm(q_bf + h * DH, qShared ? 0 : (long long)nq * V, V,
           k_bf + h * DH, (long long)nk * V, V,
           nullptr, nullptr, 0, 0,
           Sc + (size_t)h * nq * nk, (long long)H * nq * nk, nk,
           nullptr, 0, 0, nullptr, 0, 0, nq, nk, DH, BB, 0);
    // softmax rows (scale = 1/sqrt(256))
    {
      unsigned rows = (unsigned)((size_t)BB * H * nq);
      if (nk == 512) k_softmax<16><<<dim3(rows), 32, 0, stream>>>(Sc, Pr, 0.0625f);
      else           k_softmax<2> <<<dim3(rows), 32, 0, stream>>>(Sc, Pr, 0.0625f);
    }
    // O = q + P @ V : per head, B^T = vT slice, resid = q
    for (int h = 0; h < H; ++h)
      gemm(Pr + (size_t)h * nq * nk, (long long)H * nq * nk, nk,
           vT_bf + (size_t)h * DH * nk, (long long)V * nk, nk,
           nullptr,
           q_f + h * DH, qShared ? 0 : (long long)nq * V, V,
           Ob + h * DH, (long long)nq * V, V,
           nullptr, 0, 0, nullptr, 0, 0, nq, DH, nk, BB, 0);
    // ln0
    k_layernorm<<<dim3((unsigned)(BB * nq)), 32, 0, stream>>>(
        Ob, m.ln0.g, m.ln0.b, OlnF, OlnB, HIDD);
    // O2 = O_ln + relu(O_ln @ Wo + bo)
    gemm(OlnB, 0, V, tw.o, 0, V, m.o.b, OlnF, 0, V,
         O2, 0, V, nullptr, 0, 0, nullptr, 0, 0, BB * nq, V, V, 1, 1);
    // ln1
    k_layernorm<<<dim3((unsigned)(BB * nq)), 32, 0, stream>>>(
        O2, m.ln1.g, m.ln1.b, out_f, out_bf, HIDD);
  };

  // ---- encoder: two ISABs ----
  mab_big(m00, t00, I0_bf, INDSN, HIDD, true,  x_bf, NN, DIND, Hbf, nullptr);
  mab_big(m01, t01, x_bf,  NN, DIND, false, Hbf, INDSN, HIDD, tA,  nullptr);
  mab_big(m10, t10, I1_bf, INDSN, HIDD, true,  tA,  NN, HIDD, Hbf, nullptr);
  mab_big(m11, t11, tA,   NN, HIDD, false, Hbf, INDSN, HIDD, tB,  nullptr);

  // ---- PMA (nq = 1, q shared across batch) ----
  k_rowgemm<<<dim3(1, 4), 64, 0, stream>>>(Ssd, 0, mpm.q.w, mpm.q.b,
                                           nullptr, 0, qS, 0, HIDD, HIDD, 0);
  gemm(tB, 0, HIDD, pkT, 0, HIDD, mpm.k.b, nullptr, 0, 0,
       nullptr, 0, 0, k_bf, 0, HIDD, nullptr, 0, 0, BB * NN, HIDD, HIDD, 1, 0);
  gemm(tB, (long long)NN * HIDD, HIDD, pvT, 0, HIDD, mpm.v.b, nullptr, 0, 0,
       nullptr, 0, 0, nullptr, 0, 0, vT_bf, (long long)HIDD * NN, NN,
       NN, HIDD, HIDD, BB, 0);
  k_attn_q1<<<dim3(BB, NHEAD), 256, 0, stream>>>(qS, k_bf, vT_bf, Op, NN, 0.0625f);
  k_layernorm<<<dim3(BB), 32, 0, stream>>>(Op, mpm.ln0.g, mpm.ln0.b, OpL, nullptr, HIDD);
  k_rowgemm<<<dim3(BB, 4), 64, 0, stream>>>(OpL, HIDD, mpm.o.w, mpm.o.b,
                                            OpL, HIDD, Op2, HIDD, HIDD, HIDD, 1);
  k_layernorm<<<dim3(BB), 32, 0, stream>>>(Op2, mpm.ln1.g, mpm.ln1.b, tp, nullptr, HIDD);

  // ---- SABs with nq = nk = 1: softmax over one key == 1 => O = q + v ----
  auto sab = [&](const MabP& m, const float* tin, float* toutF, u16* toutB, int ldB) {
    k_rowgemm<<<dim3(BB, 4), 64, 0, stream>>>(tin, HIDD, m.q.w, m.q.b,
                                              nullptr, 0, qv, HIDD, HIDD, HIDD, 0);
    k_rowgemm<<<dim3(BB, 4), 64, 0, stream>>>(tin, HIDD, m.v.w, m.v.b,
                                              nullptr, 0, vv, HIDD, HIDD, HIDD, 0);
    k_add<<<dim3((BB * HIDD + 255) / 256), 256, 0, stream>>>(qv, vv, Os, (size_t)BB * HIDD);
    k_layernorm<<<dim3(BB), 32, 0, stream>>>(Os, m.ln0.g, m.ln0.b, OsL, nullptr, HIDD);
    k_rowgemm<<<dim3(BB, 4), 64, 0, stream>>>(OsL, HIDD, m.o.w, m.o.b,
                                              OsL, HIDD, Os2, HIDD, HIDD, HIDD, 1);
    k_layernorm<<<dim3(BB), 32, 0, stream>>>(Os2, m.ln1.g, m.ln1.b, toutF, toutB, ldB);
  };
  sab(ms0, tp, ts0, nullptr, HIDD);
  sab(ms1, ts0, nullptr, hcat, 2 * HIDD);   // t -> hcat[:, 0:256] (bf16)

  // ---- deepset branch (reuses k_bf / Ob scratch) ----
  gemm(x_bf, 0, DIND, d0T, 0, DIND, ds0.b, nullptr, 0, 0,
       nullptr, 0, 0, k_bf, 0, HIDD, nullptr, 0, 0, BB * NN, HIDD, DIND, 1, 1);
  gemm(k_bf, 0, HIDD, d1T, 0, HIDD, ds1.b, nullptr, 0, 0,
       Ob, 0, HIDD, nullptr, 0, 0, nullptr, 0, 0, BB * NN, HIDD, HIDD, 1, 1);
  k_mean<<<dim3(BB), HIDD, 0, stream>>>(Ob, NN, HIDD, hcat + HIDD, 2 * HIDD);

  // ---- head: leaky-relu linear (WMMA) + tiny final linear ----
  gemm(hcat, 0, 2 * HIDD, h0T, 0, 2 * HIDD, h0.b, nullptr, 0, 0,
       h0f, 0, HIDD / 2, nullptr, 0, 0, nullptr, 0, 0, BB, HIDD / 2, 2 * HIDD, 1, 2);
  k_rowgemm<<<dim3(BB, 1), 64, 0, stream>>>(h0f, HIDD / 2, h1.w, h1.b,
                                            nullptr, 0, (float*)d_out, 4,
                                            HIDD / 2, 4, 0);
}
